// NTXentHPS_55198919688654
// MI455X (gfx1250) — compile-verified, matrix operand order checked by reading, not run on hardware
//
#include <hip/hip_runtime.h>
#include <hip/hip_bf16.h>

typedef __attribute__((ext_vector_type(16))) _Float16     v16h;
typedef __attribute__((ext_vector_type(8)))  float        v8f;
typedef __attribute__((ext_vector_type(4)))  unsigned int v4u;

#define NROWS 16384
#define HALF  8192
#define DIMK  128
#define NTILE 512            // 8192/16 column tiles per half
#define SQRT_LOG2E 1.2011224087864498f   // sqrt(log2(e)); Gram matrix lands in log2 domain

union V16H { v16h h; struct { v4u lo; v4u hi; } u; };

// ---------------------------------------------------------------------------
// K1: row-normalize z (fp32) -> zn (f16, pre-scaled by sqrt(log2 e)), and
// store posdot = <zr_i, za_i> (fp32, natural-log domain == log(pos_sim)).
// One wave per row pair; each lane handles 4 of 128 elements.
// ---------------------------------------------------------------------------
__global__ void ntxent_norm_kernel(const float* __restrict__ z,
                                   _Float16* __restrict__ zn,
                                   float* __restrict__ posdot) {
    int pair = blockIdx.x * 8 + (threadIdx.x >> 5);   // 0..8191
    int lane = threadIdx.x & 31;

    const float4 r = *(const float4*)(z + (size_t)pair * DIMK + lane * 4);
    const float4 a = *(const float4*)(z + (size_t)(pair + HALF) * DIMK + lane * 4);

    float ssr = r.x*r.x + r.y*r.y + r.z*r.z + r.w*r.w;
    float ssa = a.x*a.x + a.y*a.y + a.z*a.z + a.w*a.w;
    float dot = r.x*a.x + r.y*a.y + r.z*a.z + r.w*a.w;

    #pragma unroll
    for (int off = 16; off >= 1; off >>= 1) {
        ssr += __shfl_xor(ssr, off, 32);
        ssa += __shfl_xor(ssa, off, 32);
        dot += __shfl_xor(dot, off, 32);
    }

    float invr = 1.0f / fmaxf(sqrtf(ssr), 1e-12f);
    float inva = 1.0f / fmaxf(sqrtf(ssa), 1e-12f);
    float sr = invr * SQRT_LOG2E;
    float sa = inva * SQRT_LOG2E;

    _Float16* pr = zn + (size_t)pair * DIMK + lane * 4;
    _Float16* pa = zn + (size_t)(pair + HALF) * DIMK + lane * 4;
    pr[0] = (_Float16)(r.x * sr); pr[1] = (_Float16)(r.y * sr);
    pr[2] = (_Float16)(r.z * sr); pr[3] = (_Float16)(r.w * sr);
    pa[0] = (_Float16)(a.x * sa); pa[1] = (_Float16)(a.y * sa);
    pa[2] = (_Float16)(a.z * sa); pa[3] = (_Float16)(a.w * sa);

    if (lane == 0) posdot[pair] = dot * invr * inva;   // log(pos_sim), TAU=1
}

// ---------------------------------------------------------------------------
// Helper: load one 16x32 A/B fragment chunk in the ISA 16-bit WMMA layout.
// ---------------------------------------------------------------------------
__device__ __forceinline__ v16h load_frag(const _Float16* row, int k0) {
    V16H t;
    t.u.lo = *(const v4u*)(row + k0);        // K = k0 .. k0+7
    t.u.hi = *(const v4u*)(row + k0 + 16);   // K = k0+16 .. k0+23
    return t.h;
}

// ---------------------------------------------------------------------------
// K2: fused Gram-row-sum: sim[row] = sum_{col != row} exp(<z_row, z_col>)
// zn is pre-scaled so the WMMA accumulator is log2(e)*dot -> use exp2.
// blockIdx.x = 32-row block (0..511; >=256 is the augmented half).
// 8 waves/block; each wave keeps TWO 16x128 A tiles (M=32) in VGPRs and
// sweeps 1/8 of the 512 column tiles, issuing 8 WMMAs per B fragment set
// (halves L2 B-traffic vs M=16 blocking). Diagonals are subtracted after the
// sweep from the register-resident A fragments (||a_row||^2); hot loop is
// mask-free.
// ---------------------------------------------------------------------------
__global__ void ntxent_sim_kernel(const _Float16* __restrict__ zn,
                                  float* __restrict__ sim) {
    const int blk     = blockIdx.x;            // 0..511
    const int halfIdx = blk >> 8;              // 0 or 1
    const int rowBase = halfIdx * HALF + (blk & 255) * 32;   // global row index
    const int lane    = threadIdx.x & 31;
    const int wave    = threadIdx.x >> 5;      // 0..7
    const int n16     = lane & 15;
    const int hs      = (lane >> 4) << 3;      // 0 for lanes 0-15, 8 for 16-31

    // --- load two A row-tiles (ISA 16-bit 16x32 layout), kept for whole sweep
    const _Float16* arow0 = zn + (size_t)(rowBase + n16) * DIMK;
    const _Float16* arow1 = zn + (size_t)(rowBase + 16 + n16) * DIMK;
    v16h A0[4], A1[4];
    #pragma unroll
    for (int c = 0; c < 4; ++c) {
        A0[c] = load_frag(arow0, 32 * c + hs);
        A1[c] = load_frag(arow1, 32 * c + hs);
    }

    float rsum0[8] = {0.f, 0.f, 0.f, 0.f, 0.f, 0.f, 0.f, 0.f};
    float rsum1[8] = {0.f, 0.f, 0.f, 0.f, 0.f, 0.f, 0.f, 0.f};

    for (int ct = wave; ct < NTILE; ct += 8) {
        const _Float16* brow = zn + (size_t)(halfIdx * HALF + ct * 16 + n16) * DIMK;

        v8f C0 = {}, C1 = {};
        #pragma unroll
        for (int c = 0; c < 4; ++c) {
            v16h B = load_frag(brow, 32 * c + hs);
            C0 = __builtin_amdgcn_wmma_f32_16x16x32_f16(
                     false, A0[c], false, B, (short)0, C0, false, false);
            C1 = __builtin_amdgcn_wmma_f32_16x16x32_f16(
                     false, A1[c], false, B, (short)0, C1, false, false);
        }

        // C[v] = log2(e) * S[M][N], M = v + hs (+16 for tile 1), N = n16
        #pragma unroll
        for (int v = 0; v < 8; ++v) {
            rsum0[v] += __builtin_amdgcn_exp2f(C0[v]);
            rsum1[v] += __builtin_amdgcn_exp2f(C1[v]);
        }
    }

    // --- remove diagonal terms exp(<row,row>) once, from registers.
    // Lanes L and L+16 jointly hold the 128 K-values of rows n16 / n16+16.
    float ssq0 = 0.f, ssq1 = 0.f;
    #pragma unroll
    for (int c = 0; c < 4; ++c)
        #pragma unroll
        for (int j = 0; j < 16; ++j) {
            float a0 = (float)A0[c][j]; ssq0 += a0 * a0;
            float a1 = (float)A1[c][j]; ssq1 += a1 * a1;
        }
    ssq0 += __shfl_xor(ssq0, 16, 32);           // combine the two K-halves
    ssq1 += __shfl_xor(ssq1, 16, 32);
    float dE0 = __builtin_amdgcn_exp2f(ssq0);   // = exp(||z_row||^2 / TAU)
    float dE1 = __builtin_amdgcn_exp2f(ssq1);
    #pragma unroll
    for (int v = 0; v < 8; ++v)
        if (n16 == v + hs) { rsum0[v] -= dE0; rsum1[v] -= dE1; }

    // reduce over N: columns live across the 16 lanes of each half-group
    #pragma unroll
    for (int v = 0; v < 8; ++v)
        #pragma unroll
        for (int off = 8; off >= 1; off >>= 1) {
            rsum0[v] += __shfl_xor(rsum0[v], off, 16);
            rsum1[v] += __shfl_xor(rsum1[v], off, 16);
        }

    __shared__ float srow[32];
    if (threadIdx.x < 32) srow[threadIdx.x] = 0.f;
    __syncthreads();
    if (n16 == 0) {      // lanes 0 and 16 hold totals for M = v and v+8
        #pragma unroll
        for (int v = 0; v < 8; ++v) {
            atomicAdd(&srow[v + hs],      rsum0[v]);
            atomicAdd(&srow[16 + v + hs], rsum1[v]);
        }
    }
    __syncthreads();
    if (threadIdx.x < 32)
        sim[rowBase + threadIdx.x] = srow[threadIdx.x];
}

// ---------------------------------------------------------------------------
// K3: loss = -mean( posdot[i] - log(sim_real[i]) - log(sim_aug[i]) )
// ---------------------------------------------------------------------------
__global__ void ntxent_loss_kernel(const float* __restrict__ posdot,
                                   const float* __restrict__ sim,
                                   float* __restrict__ out) {
    __shared__ float red[256];
    float s = 0.f;
    for (int i = threadIdx.x; i < HALF; i += 256)
        s += posdot[i] - __logf(sim[i]) - __logf(sim[i + HALF]);
    red[threadIdx.x] = s;
    __syncthreads();
    #pragma unroll
    for (int stride = 128; stride > 0; stride >>= 1) {
        if (threadIdx.x < stride) red[threadIdx.x] += red[threadIdx.x + stride];
        __syncthreads();
    }
    if (threadIdx.x == 0) {
        out[0] = -red[0] / (float)HALF;
        out[1] = 0.f;
    }
}

// ---------------------------------------------------------------------------
extern "C" void kernel_launch(void* const* d_in, const int* in_sizes, int n_in,
                              void* d_out, int out_size, void* d_ws, size_t ws_size,
                              hipStream_t stream) {
    (void)in_sizes; (void)n_in; (void)out_size; (void)ws_size;

    const float* z = (const float*)d_in[0];
    float* out     = (float*)d_out;

    // workspace layout
    char* ws = (char*)d_ws;
    _Float16* zn   = (_Float16*)ws;                                  // 4 MB
    float* posdot  = (float*)(ws + (size_t)NROWS * DIMK * 2);        // 32 KB
    float* sim     = posdot + HALF;                                  // 64 KB

    ntxent_norm_kernel<<<dim3(HALF / 8), dim3(256), 0, stream>>>(z, zn, posdot);
    ntxent_sim_kernel <<<dim3(512),      dim3(256), 0, stream>>>(zn, sim);
    ntxent_loss_kernel<<<dim3(1),        dim3(256), 0, stream>>>(posdot, sim, out);
}
